// CustomSelfAttention_24257975288159
// MI455X (gfx1250) — matrix operand order repaired and checked by reading, not compiled, not run on hardware
//
#include <hip/hip_runtime.h>

// ---------------------------------------------------------------------------
// Problem constants
// ---------------------------------------------------------------------------
#define BATCH 4
#define SEQ   2048
#define DIM   1024
#define HEADS 16
#define HD    64
#define NROWS (BATCH * SEQ)        // 8192
#define ATT_SCALE 0.125f           // 1/sqrt(64)

typedef __attribute__((ext_vector_type(16))) __bf16 v16bf;
typedef __attribute__((ext_vector_type(8)))  __bf16 v8bf;
typedef __attribute__((ext_vector_type(8)))  float  v8f;

// ---------------------------------------------------------------------------
// WMMA helpers (CDNA5 16x16x32 bf16, f32 accumulate)
// ---------------------------------------------------------------------------
__device__ __forceinline__ v16bf cat8(v8bf lo, v8bf hi) {
  return __builtin_shufflevector(lo, hi, 0,1,2,3,4,5,6,7,8,9,10,11,12,13,14,15);
}

// A operand (16x32, MxK, 16-bit): lanes 0-15 hold row M=lane, K = {0..7,16..23};
// lanes 16-31 hold row M=lane-16, K = {8..15,24..31}.
__device__ __forceinline__ v16bf load_a16(const __bf16* row, int k0, int lane) {
  int koff = (lane & 16) ? 8 : 0;
  v8bf lo = *(const v8bf*)(row + k0 + koff);
  v8bf hi = *(const v8bf*)(row + k0 + koff + 16);
  return cat8(lo, hi);
}

// B operand (32x16, KxN, 16-bit): lanes 0-15 hold K = {0..7,16..23} of column N=lane,
// lanes 16-31 hold K = {8..15,24..31} of column N=lane-16. `col` points at a contiguous
// K-major run for one column (we store W / K / V so this is contiguous).
__device__ __forceinline__ v16bf load_b16(const __bf16* col, int k0, int lane) {
  int koff = (lane & 16) ? 8 : 0;
  v8bf lo = *(const v8bf*)(col + k0 + koff);
  v8bf hi = *(const v8bf*)(col + k0 + koff + 16);
  return cat8(lo, hi);
}

__device__ __forceinline__ v8f wmma_bf16(v16bf a, v16bf b, v8f c) {
  return __builtin_amdgcn_wmma_f32_16x16x32_bf16(
      /*neg_a=*/false, a, /*neg_b=*/false, b,
      /*c_mod=*/(short)0, c, /*reuse_a=*/false, /*reuse_b=*/false);
}

// ---------------------------------------------------------------------------
// 1) Convert x -> bf16 (plain copy + copy with CLS-sent row-1 substitution)
// ---------------------------------------------------------------------------
__global__ __launch_bounds__(256) void convert_x_kernel(
    const float* __restrict__ x, const float* __restrict__ cross,
    __bf16* __restrict__ x_bf, __bf16* __restrict__ xq_bf) {
  size_t i = (size_t)blockIdx.x * 256 + threadIdx.x;   // < 8388608
  float v = x[i];
  x_bf[i] = (__bf16)v;
  size_t d = i & (DIM - 1);
  size_t l = (i >> 10) & (SEQ - 1);
  size_t b = i >> 21;
  xq_bf[i] = (__bf16)((l == 1) ? cross[b * DIM + d] : v);
}

// 2) Convert+transpose weights: Wt[n][k] = W[k][n], bf16
__global__ __launch_bounds__(256) void convert_wt_kernel(
    const float* __restrict__ W, __bf16* __restrict__ Wt) {
  size_t i = (size_t)blockIdx.x * 256 + threadIdx.x;   // i = k*DIM + n
  size_t k = i >> 10, n = i & (DIM - 1);
  Wt[n * DIM + k] = (__bf16)W[i];
}

// ---------------------------------------------------------------------------
// 3) Projection GEMM: out = A(8192x1024) * Wt^T + bias, head-major layouts.
//    Wave tile = 32 (M) x 64 (N): 8 WMMAs per 32-deep k-step against 6 loads.
//    mode 0: out[((b*H+h)*SEQ + l)*HD + d]   (Q, K)
//    mode 1: out[((b*H+h)*HD + d)*SEQ + l]   (V, transposed for PV B-operand)
// ---------------------------------------------------------------------------
__global__ __launch_bounds__(256) void gemm_qkv_kernel(
    const __bf16* __restrict__ A, const __bf16* __restrict__ Wt,
    const float* __restrict__ bias, __bf16* __restrict__ out,
    int mode, float scale) {
  int lane = threadIdx.x & 31;
  int wid  = threadIdx.x >> 5;
  int m0 = blockIdx.x * 32;                 // 256 row tiles of 32
  int n0 = blockIdx.y * 512 + wid * 64;     // 2 col blocks * 8 waves * 64 cols

  const __bf16* arow0 = A + (size_t)(m0 + (lane & 15)) * DIM;
  const __bf16* arow1 = arow0 + (size_t)16 * DIM;

  v8f acc[2][4] = {};
  for (int k0 = 0; k0 < DIM; k0 += 32) {
    if (k0 + 256 < DIM) {
      __builtin_prefetch(arow0 + k0 + 256, 0, 3);   // WGP-scope prefetch
      __builtin_prefetch(arow1 + k0 + 256, 0, 3);
    }
    v16bf a0 = load_a16(arow0, k0, lane);
    v16bf a1 = load_a16(arow1, k0, lane);
#pragma unroll
    for (int t = 0; t < 4; ++t) {
      const __bf16* wcol = Wt + (size_t)(n0 + t * 16 + (lane & 15)) * DIM;
      if (k0 + 256 < DIM) __builtin_prefetch(wcol + k0 + 256, 0, 3);
      v16bf b = load_b16(wcol, k0, lane);
      acc[0][t] = wmma_bf16(a0, b, acc[0][t]);
      acc[1][t] = wmma_bf16(a1, b, acc[1][t]);
    }
  }

  int bb = m0 >> 11;              // batch (SEQ=2048 rows per batch)
  int l0 = m0 & (SEQ - 1);
  int rbase = (lane & 16) ? 8 : 0;
#pragma unroll
  for (int t = 0; t < 4; ++t) {
    int ng = n0 + t * 16 + (lane & 15);
    float bval = bias[ng];
    int h = ng >> 6, d = ng & 63;
#pragma unroll
    for (int g = 0; g < 2; ++g) {
#pragma unroll
      for (int r = 0; r < 8; ++r) {
        int l = l0 + g * 16 + rbase + r;
        float v = (acc[g][t][r] + bval) * scale;
        size_t idx;
        if (mode == 0)
          idx = (((size_t)(bb * HEADS + h) * SEQ) + l) * HD + d;
        else
          idx = (((size_t)(bb * HEADS + h) * HD) + d) * SEQ + l;
        out[idx] = (__bf16)v;
      }
    }
  }
}

// ---------------------------------------------------------------------------
// 4) Flash attention: per wave one 16-row q tile; online softmax over 32-key
//    chunks; LDS transpose of P (C-layout -> A-operand layout).
// ---------------------------------------------------------------------------
__global__ __launch_bounds__(256) void attn_kernel(
    const __bf16* __restrict__ Qh, const __bf16* __restrict__ Kh,
    const __bf16* __restrict__ Vt, const float* __restrict__ mask,
    float* __restrict__ out) {
  __shared__ __attribute__((aligned(32))) __bf16 lds_p[8][16][32];

  int lane = threadIdx.x & 31;
  int wid  = threadIdx.x >> 5;
  int bh   = blockIdx.x >> 4;          // 0..63  (b*H + h)
  int tile = blockIdx.x & 15;          // 16 q-tiles of 128 rows
  int b = bh >> 4, h = bh & 15;
  int q0 = tile * 128 + wid * 16;

  const __bf16* Qb = Qh + (size_t)bh * SEQ * HD;
  const __bf16* Kb = Kh + (size_t)bh * SEQ * HD;
  const __bf16* Vb = Vt + (size_t)bh * HD * SEQ;
  const float*  mrow = mask + (size_t)b * SEQ;

  // Q tile A-operands (Q already scaled by 1/sqrt(HD) in projection)
  const __bf16* qrow = Qb + (size_t)(q0 + (lane & 15)) * HD;
  v16bf aq0 = load_a16(qrow, 0, lane);
  v16bf aq1 = load_a16(qrow, 32, lane);

  v8f o[4] = {};
  float m_i[8], l_i[8];
#pragma unroll
  for (int r = 0; r < 8; ++r) { m_i[r] = -INFINITY; l_i[r] = 0.f; }

  int rbase = (lane & 16) ? 8 : 0;

  for (int kt = 0; kt < SEQ; kt += 32) {
    // prefetch next chunk's K rows / V columns (WGP scope)
    if (kt + 32 < SEQ) {
      __builtin_prefetch(Kb + (size_t)(kt + 32 + (lane & 15)) * HD, 0, 3);
      __builtin_prefetch(Vb + (size_t)(lane & 15) * SEQ + kt + 32, 0, 3);
    }
    // ---- scores for two 16-key tiles (contraction over HD=64 = 2 WMMAs each)
    v8f s0 = {}, s1 = {};
    {
      const __bf16* kcol0 = Kb + (size_t)(kt + (lane & 15)) * HD;
      v16bf b0 = load_b16(kcol0, 0, lane);
      v16bf b1 = load_b16(kcol0, 32, lane);
      s0 = wmma_bf16(aq0, b0, s0);
      s0 = wmma_bf16(aq1, b1, s0);
      const __bf16* kcol1 = Kb + (size_t)(kt + 16 + (lane & 15)) * HD;
      b0 = load_b16(kcol1, 0, lane);
      b1 = load_b16(kcol1, 32, lane);
      s1 = wmma_bf16(aq0, b0, s1);
      s1 = wmma_bf16(aq1, b1, s1);
    }
    float mk0 = mrow[kt + (lane & 15)];
    float mk1 = mrow[kt + 16 + (lane & 15)];

    // ---- online softmax (row-wise across 16 lanes of each half-wave)
#pragma unroll
    for (int r = 0; r < 8; ++r) {
      float v0 = s0[r] + mk0;
      float v1 = s1[r] + mk1;
      float tmax = fmaxf(v0, v1);
#pragma unroll
      for (int mm = 1; mm < 16; mm <<= 1)
        tmax = fmaxf(tmax, __shfl_xor(tmax, mm, 32));
      float nm   = fmaxf(m_i[r], tmax);
      float corr = __expf(m_i[r] - nm);
      float p0   = __expf(v0 - nm);
      float p1   = __expf(v1 - nm);
      float rs   = p0 + p1;
#pragma unroll
      for (int mm = 1; mm < 16; mm <<= 1)
        rs += __shfl_xor(rs, mm, 32);
      l_i[r] = l_i[r] * corr + rs;
      m_i[r] = nm;
      o[0][r] *= corr; o[1][r] *= corr; o[2][r] *= corr; o[3][r] *= corr;
      // stash P in LDS (row-major 16x32 bf16) for layout transpose
      lds_p[wid][rbase + r][lane & 15]        = (__bf16)p0;
      lds_p[wid][rbase + r][16 + (lane & 15)] = (__bf16)p1;
    }

    // wave-private LDS region; cross-lane readback needs DS counter drained
    asm volatile("s_wait_dscnt 0" ::: "memory");

    // ---- read P back in A-operand layout
    {
      int row  = lane & 15;
      int koff = (lane & 16) ? 8 : 0;
      v8bf lo = *(const v8bf*)&lds_p[wid][row][koff];
      v8bf hi = *(const v8bf*)&lds_p[wid][row][16 + koff];
      v16bf ap = cat8(lo, hi);
      // ---- PV: contraction over 32 keys, N = HD in 4 tiles of 16
#pragma unroll
      for (int t = 0; t < 4; ++t) {
        const __bf16* vcol = Vb + (size_t)(t * 16 + (lane & 15)) * SEQ;
        v16bf bv = load_b16(vcol, kt, lane);
        o[t] = wmma_bf16(ap, bv, o[t]);
      }
    }
  }

  // ---- normalize and store f32 output in (B, L, D) layout
  float invl[8];
#pragma unroll
  for (int r = 0; r < 8; ++r) invl[r] = 1.f / l_i[r];
#pragma unroll
  for (int t = 0; t < 4; ++t) {
#pragma unroll
    for (int r = 0; r < 8; ++r) {
      int l = q0 + rbase + r;
      int col = h * HD + t * 16 + (lane & 15);
      out[((size_t)b * SEQ + l) * DIM + col] = o[t][r] * invl[r];
    }
  }
}

// ---------------------------------------------------------------------------
// Launch
// ---------------------------------------------------------------------------
extern "C" void kernel_launch(void* const* d_in, const int* in_sizes, int n_in,
                              void* d_out, int out_size, void* d_ws, size_t ws_size,
                              hipStream_t stream) {
  const float* x     = (const float*)d_in[0];
  const float* mask  = (const float*)d_in[1];
  const float* cross = (const float*)d_in[2];
  const float* Wq    = (const float*)d_in[3];
  const float* bq    = (const float*)d_in[4];
  const float* Wk    = (const float*)d_in[5];
  const float* bk    = (const float*)d_in[6];
  const float* Wv    = (const float*)d_in[7];
  const float* bv    = (const float*)d_in[8];
  float* out = (float*)d_out;

  char* ws = (char*)d_ws;
  const size_t XB   = (size_t)NROWS * DIM * sizeof(__bf16);   // 16 MB
  const size_t WB   = (size_t)DIM * DIM * sizeof(__bf16);     // 2 MB
  __bf16* x_bf  = (__bf16*)(ws);
  __bf16* xq_bf = (__bf16*)(ws + XB);
  __bf16* wq_t  = (__bf16*)(ws + 2 * XB);
  __bf16* wk_t  = (__bf16*)(ws + 2 * XB + WB);
  __bf16* wv_t  = (__bf16*)(ws + 2 * XB + 2 * WB);
  __bf16* Qh    = (__bf16*)(ws + 2 * XB + 3 * WB);
  __bf16* Kh    = (__bf16*)(ws + 3 * XB + 3 * WB);
  __bf16* Vt    = (__bf16*)(ws + 4 * XB + 3 * WB);

  convert_x_kernel<<<(NROWS * DIM) / 256, 256, 0, stream>>>(x, cross, x_bf, xq_bf);
  convert_wt_kernel<<<(DIM * DIM) / 256, 256, 0, stream>>>(Wq, wq_t);
  convert_wt_kernel<<<(DIM * DIM) / 256, 256, 0, stream>>>(Wk, wk_t);
  convert_wt_kernel<<<(DIM * DIM) / 256, 256, 0, stream>>>(Wv, wv_t);

  dim3 ggrid(NROWS / 32, DIM / 512);
  gemm_qkv_kernel<<<ggrid, 256, 0, stream>>>(xq_bf, wq_t, bq, Qh, 0, ATT_SCALE);
  gemm_qkv_kernel<<<ggrid, 256, 0, stream>>>(x_bf,  wk_t, bk, Kh, 0, 1.0f);
  gemm_qkv_kernel<<<ggrid, 256, 0, stream>>>(x_bf,  wv_t, bv, Vt, 1, 1.0f);

  attn_kernel<<<BATCH * HEADS * (SEQ / 128), 256, 0, stream>>>(Qh, Kh, Vt, mask, out);
}
